// BinaryMoSLinear_28982439313533
// MI455X (gfx1250) — matrix operand order repaired and checked by reading, not compiled, not run on hardware
//
#include <hip/hip_runtime.h>
#include <hip/hip_fp16.h>

typedef __attribute__((ext_vector_type(16))) _Float16 v16h;
typedef __attribute__((ext_vector_type(8)))  float    v8f;

#define H_DIM  4096
#define O_DIM  4096
#define N_ROWS 8192
#define N_EXP  8

union FragU { uint4 u[2]; v16h h; };

// ---------------------------------------------------------------------------
// Async copy of 64 contiguous bytes (4 x b128) global -> LDS, per lane.
// INST_OFFSET is added to both the LDS and global addresses (ISA 08 §4.4),
// so one LDS base + one global base covers all four transfers.
// Tracked by ASYNCcnt (8 ops/thread per K-tile: 4 for A + 4 for B).
// ---------------------------------------------------------------------------
__device__ __forceinline__ void async_copy64B(unsigned lds_off, const void* gptr) {
    asm volatile(
        "global_load_async_to_lds_b128 %0, %1, off\n\t"
        "global_load_async_to_lds_b128 %0, %1, off offset:16\n\t"
        "global_load_async_to_lds_b128 %0, %1, off offset:32\n\t"
        "global_load_async_to_lds_b128 %0, %1, off offset:48"
        :: "v"(lds_off), "v"(gptr) : "memory");
}

// ---------------------------------------------------------------------------
// Kernel 1: per-row router logits -> softmax -> in_scale fuse -> f16 activations
// ---------------------------------------------------------------------------
__global__ __launch_bounds__(256)
void prep_kernel(const float* __restrict__ x,
                 const float* __restrict__ gate_w,
                 const float* __restrict__ ics,
                 _Float16* __restrict__ a_scaled,
                 float* __restrict__ routing) {
    const int n    = blockIdx.x;
    const int tid  = threadIdx.x;
    const int lane = tid & 31;
    const int wave = tid >> 5;          // 0..7 == expert id

    __shared__ float s_logit[N_EXP];

    const float* xr = x + (size_t)n * H_DIM;

    // ---- phase 1: logit[e] = dot(x_row, gate_w[e]) ----
    {
        const float* gr = gate_w + (size_t)wave * H_DIM;
        float p = 0.f;
        for (int k = lane; k < H_DIM; k += 32)
            p += xr[k] * gr[k];
        #pragma unroll
        for (int off = 16; off > 0; off >>= 1)
            p += __shfl_xor(p, off, 32);
        if (lane == 0) s_logit[wave] = p;
    }
    __syncthreads();

    // ---- softmax over 8 experts (redundant per thread, cheap) ----
    float r[N_EXP];
    float m = s_logit[0];
    #pragma unroll
    for (int e = 1; e < N_EXP; ++e) m = fmaxf(m, s_logit[e]);
    float sum = 0.f;
    #pragma unroll
    for (int e = 0; e < N_EXP; ++e) { r[e] = __expf(s_logit[e] - m); sum += r[e]; }
    const float inv = 1.f / sum;
    #pragma unroll
    for (int e = 0; e < N_EXP; ++e) r[e] *= inv;

    if (tid < N_EXP) routing[(size_t)n * N_EXP + tid] = r[tid];

    // ---- phase 2: a_scaled[n,h] = f16( x[n,h] * sum_e r[e]*ics[e,h] ) ----
    for (int h0 = tid * 2; h0 < H_DIM; h0 += 512) {
        float s0 = 0.f, s1 = 0.f;
        #pragma unroll
        for (int e = 0; e < N_EXP; ++e) {
            const float* ic = ics + (size_t)e * H_DIM + h0;
            s0 += r[e] * ic[0];
            s1 += r[e] * ic[1];
        }
        union { _Float16 h[2]; unsigned int u; } pk;
        pk.h[0] = (_Float16)(xr[h0]     * s0);
        pk.h[1] = (_Float16)(xr[h0 + 1] * s1);
        *(unsigned int*)(a_scaled + (size_t)n * H_DIM + h0) = pk.u;
    }
}

// ---------------------------------------------------------------------------
// Kernel 2: w_sign = sign(weight) as f16 (+1.0 / -1.0 / 0.0), bit-selected
// ---------------------------------------------------------------------------
__global__ __launch_bounds__(256)
void sign_kernel(const float* __restrict__ w, unsigned int* __restrict__ out) {
    const size_t i = (size_t)blockIdx.x * 256 + threadIdx.x;   // float4 index
    const float4 v = ((const float4*)w)[i];
    unsigned short s0 = v.x > 0.f ? 0x3C00u : (v.x < 0.f ? 0xBC00u : 0u);
    unsigned short s1 = v.y > 0.f ? 0x3C00u : (v.y < 0.f ? 0xBC00u : 0u);
    unsigned short s2 = v.z > 0.f ? 0x3C00u : (v.z < 0.f ? 0xBC00u : 0u);
    unsigned short s3 = v.w > 0.f ? 0x3C00u : (v.w < 0.f ? 0xBC00u : 0u);
    out[i * 2 + 0] = (unsigned int)s0 | ((unsigned int)s1 << 16);
    out[i * 2 + 1] = (unsigned int)s2 | ((unsigned int)s3 << 16);
}

// ---------------------------------------------------------------------------
// Kernel 3: 128x128 WMMA GEMM, double-buffered LDS fed by async global->LDS
// copies (ASYNCcnt), fused out_scale/bias epilogue.
// 256 threads = 8 waves (4 M x 2 N); each wave -> 32x64 output (2x4 tiles).
// LDS row stride padded to 144 B (9 x uint4): 16B-aligned, bank-spread.
// ---------------------------------------------------------------------------
__global__ __launch_bounds__(256)
void gemm_kernel(const _Float16* __restrict__ A,     // [N_ROWS, H]
                 const _Float16* __restrict__ Wb,    // [O, H]  (±1 in f16)
                 const float* __restrict__ routing,  // [N_ROWS, 8]
                 const float* __restrict__ ocs,      // [8, O]
                 const float* __restrict__ bias,     // [O]
                 float* __restrict__ out) {          // [N_ROWS, O]
    __shared__ uint4 sA[2][128 * 9];     // 2 x (128 rows x 64 f16, padded)
    __shared__ uint4 sB[2][128 * 9];
    __shared__ float s_r[128][N_EXP];
    __shared__ float s_ocs[N_EXP][128];
    __shared__ float s_bias[128];

    const int tid  = threadIdx.x;
    const int gm0  = blockIdx.y * 128;   // output row base
    const int gn0  = blockIdx.x * 128;   // output col base

    // stage epilogue operands
    for (int i = tid; i < 128 * N_EXP; i += 256)
        ((float*)s_r)[i] = routing[(size_t)gm0 * N_EXP + i];      // [row][e] contiguous
    for (int i = tid; i < N_EXP * 128; i += 256)
        s_ocs[i >> 7][i & 127] = ocs[(size_t)(i >> 7) * O_DIM + gn0 + (i & 127)];
    if (tid < 128) s_bias[tid] = bias[gn0 + tid];

    const int lrow  = tid >> 1;          // 0..127 (tile row this thread stages)
    const int lhalf = tid & 1;           // which 64B half of the 128B row
    const int wid   = tid >> 5;
    const int lane  = tid & 31;
    const int wm    = (wid & 3) * 32;    // wave's M base within block
    const int wn    = (wid >> 2) * 64;   // wave's N base within block
    const int l16   = lane & 15;
    const int hiL   = lane >> 4;         // 0 for lanes 0-15, 1 for 16-31

    // per-thread staging addresses (advance K by 64 elements per tile)
    const _Float16* pA = A  + (size_t)(gm0 + lrow) * H_DIM + lhalf * 32;
    const _Float16* pB = Wb + (size_t)(gn0 + lrow) * H_DIM + lhalf * 32;
    const int      sIdx  = lrow * 9 + lhalf * 4;
    const unsigned ldsA0 = (unsigned)(size_t)&sA[0][sIdx];
    const unsigned ldsA1 = (unsigned)(size_t)&sA[1][sIdx];
    const unsigned ldsB0 = (unsigned)(size_t)&sB[0][sIdx];
    const unsigned ldsB1 = (unsigned)(size_t)&sB[1][sIdx];

    // prologue: async-fill buffer 0 with K-tile 0
    async_copy64B(ldsA0, pA);
    async_copy64B(ldsB0, pB);

    v8f acc[2][4] = {};

    constexpr int NT = H_DIM / 64;       // 64 K-tiles
    for (int t = 0; t < NT; ++t) {
        const int cur = t & 1;
        if (t + 1 < NT) {
            // issue next tile into the other buffer (safe: everyone finished
            // reading it at the barrier that ended iteration t-1)
            const _Float16* nA = pA + (size_t)(t + 1) * 64;
            const _Float16* nB = pB + (size_t)(t + 1) * 64;
            async_copy64B(cur ? ldsA0 : ldsA1, nA);
            async_copy64B(cur ? ldsB0 : ldsB1, nB);
            // 16 outstanding -> wait until only the 8 just-issued remain:
            // current tile's copies have landed.
            asm volatile("s_wait_asynccnt 0x8" ::: "memory");
        } else {
            asm volatile("s_wait_asynccnt 0x0" ::: "memory");
        }
        __syncthreads();                 // all waves' copies visible

        const uint4* tA = sA[cur];
        const uint4* tB = sB[cur];
        #pragma unroll
        for (int kk = 0; kk < 64; kk += 32) {
            // A 16x32 f16 frag: lanes 0-15 -> K {0..7,16..23}, 16-31 -> {8..15,24..31}
            FragU af[2];
            #pragma unroll
            for (int sm = 0; sm < 2; ++sm) {
                const int row  = wm + sm * 16 + l16;
                const int base = row * 9 + (kk >> 3) + hiL;
                af[sm].u[0] = tA[base];
                af[sm].u[1] = tA[base + 2];
            }
            // B 32x16 f16 frag: lanes 0-15 -> K {0..7,16..23}, 16-31 -> {8..15,24..31}
            FragU bf[4];
            #pragma unroll
            for (int sn = 0; sn < 4; ++sn) {
                const int col  = wn + sn * 16 + l16;
                const int base = col * 9 + (kk >> 3) + hiL;
                bf[sn].u[0] = tB[base];
                bf[sn].u[1] = tB[base + 2];
            }
            #pragma unroll
            for (int sm = 0; sm < 2; ++sm)
                #pragma unroll
                for (int sn = 0; sn < 4; ++sn)
                    acc[sm][sn] = __builtin_amdgcn_wmma_f32_16x16x32_f16(
                        false, af[sm].h, false, bf[sn].h,
                        (short)0, acc[sm][sn], false, false);
        }
        __syncthreads();                 // tile fully consumed before refill
    }

    // epilogue: C VGPR i, lanes 0-15 -> (M=i, N=lane), lanes 16-31 -> (M=i+8, N=lane-16)
    #pragma unroll
    for (int sm = 0; sm < 2; ++sm) {
        const int rbase = wm + sm * 16 + hiL * 8;
        #pragma unroll
        for (int sn = 0; sn < 4; ++sn) {
            const int col = wn + sn * 16 + l16;
            float oc[N_EXP];
            #pragma unroll
            for (int e = 0; e < N_EXP; ++e) oc[e] = s_ocs[e][col];
            const float bcol = s_bias[col];
            #pragma unroll
            for (int i = 0; i < 8; ++i) {
                const int row = rbase + i;
                float os = 0.f;
                #pragma unroll
                for (int e = 0; e < N_EXP; ++e) os += s_r[row][e] * oc[e];
                out[(size_t)(gm0 + row) * O_DIM + gn0 + col] = acc[sm][sn][i] * os + bcol;
            }
        }
    }
}

// ---------------------------------------------------------------------------
extern "C" void kernel_launch(void* const* d_in, const int* in_sizes, int n_in,
                              void* d_out, int out_size, void* d_ws, size_t ws_size,
                              hipStream_t stream) {
    const float* x      = (const float*)d_in[0];  // [4,2048,4096]
    const float* weight = (const float*)d_in[1];  // [4096,4096]
    const float* bias   = (const float*)d_in[2];  // [4096]
    const float* gate_w = (const float*)d_in[3];  // [8,4096]
    const float* ics    = (const float*)d_in[4];  // [8,4096]
    const float* ocs    = (const float*)d_in[5];  // [8,4096]
    float* out = (float*)d_out;

    char* ws = (char*)d_ws;
    _Float16* a_scaled = (_Float16*)ws;                                   // 64 MB
    _Float16* w_sign   = (_Float16*)(ws + (size_t)N_ROWS * H_DIM * 2);    // 32 MB
    float*    routing  = (float*)(ws + (size_t)N_ROWS * H_DIM * 2
                                     + (size_t)O_DIM  * H_DIM * 2);       // 256 KB

    prep_kernel<<<N_ROWS, 256, 0, stream>>>(x, gate_w, ics, a_scaled, routing);

    sign_kernel<<<(O_DIM * H_DIM / 4) / 256, 256, 0, stream>>>(
        weight, (unsigned int*)w_sign);

    dim3 grid(O_DIM / 128, N_ROWS / 128);   // 32 x 64 blocks
    gemm_kernel<<<grid, 256, 0, stream>>>(a_scaled, w_sign, routing, ocs, bias, out);
}
